// SegmentFCN_74466142978585
// MI455X (gfx1250) — matrix operand order repaired and checked by reading, not compiled, not run on hardware
//
#include <hip/hip_runtime.h>

typedef __attribute__((ext_vector_type(16))) __bf16        v16bf;
typedef __attribute__((ext_vector_type(8)))  float         v8f;
typedef __attribute__((ext_vector_type(4)))  unsigned int  v4u;

#define NLAB   33
#define BATCH  64
#define DMODEL 128
#define VWIDTH 902629

// LDS A-tile: bf16, pre-swizzled to WMMA fragment order.
// Layout: [mb(4)][lane(32)][16 bf16 = 8 dwords, padded to 12 dwords]
#define A_LANE_STRIDE 12                     // dwords; 48B -> 16B aligned, low conflict
#define A_LDS_DWORDS  (4 * 32 * A_LANE_STRIDE)

struct Q8 { v4u lo; v4u hi; };               // 32 bytes == v16bf

struct L1Args {
    const int*   idx[NLAB];
    const float* w1[NLAB];
    const float* b1[NLAB];
    int          n[NLAB];
};

// ---------------------------------------------------------------------------
// A 16x32 bf16 layout chosen here (contiguous k, B matches): lane = (r&15) | ((k>>4&1)<<4),
//                           elem = (k&15)
// ---------------------------------------------------------------------------

// Stage one (even,odd) pair (guarded tail path).
__device__ __forceinline__ void stage_pair(unsigned int* AswU, int r, int c,
                                           float f0, float f1) {
    int mb   = r >> 4;
    int lane = (r & 15) | (((c >> 4) & 1) << 4);
    int e    = (c & 15);                                  // even
    unsigned int u0 = (unsigned int)__builtin_bit_cast(unsigned short, (__bf16)f0);
    unsigned int u1 = (unsigned int)__builtin_bit_cast(unsigned short, (__bf16)f1);
    AswU[(mb * 32 + lane) * A_LANE_STRIDE + (e >> 1)] = u0 | (u1 << 16);
}

// Stage a quad (c % 4 == 0): 2 packed dwords, one 8-byte LDS store.
// For c%4==0 the 4 K-values map to 4 consecutive fragment elements.
__device__ __forceinline__ void stage_quad(unsigned int* AswU, int r, int c,
                                           float f0, float f1, float f2, float f3) {
    int mb   = r >> 4;
    int lane = (r & 15) | (((c >> 4) & 1) << 4);
    int e    = (c & 15);                                  // multiple of 4
    unsigned int u0 = (unsigned int)__builtin_bit_cast(unsigned short, (__bf16)f0);
    unsigned int u1 = (unsigned int)__builtin_bit_cast(unsigned short, (__bf16)f1);
    unsigned int u2 = (unsigned int)__builtin_bit_cast(unsigned short, (__bf16)f2);
    unsigned int u3 = (unsigned int)__builtin_bit_cast(unsigned short, (__bf16)f3);
    uint2 d;
    d.x = u0 | (u1 << 16);
    d.y = u2 | (u3 << 16);
    *(uint2*)(AswU + (mb * 32 + lane) * A_LANE_STRIDE + (e >> 1)) = d;   // 8B aligned
}

// Read a full A fragment: two 16-byte LDS loads, no conversion.
__device__ __forceinline__ v16bf read_a_frag(const unsigned int* AswU, int lane, int mb) {
    const v4u* p = (const v4u*)(AswU + (mb * 32 + lane) * A_LANE_STRIDE);
    Q8 q;
    q.lo = p[0];
    q.hi = p[1];
    return __builtin_bit_cast(v16bf, q);
}

// Build B fragment (16 weights of row n, K-range [k0+kh, k0+kh+16)) -> bf16.
__device__ __forceinline__ v16bf make_b_frag(const float* __restrict__ W,
                                             long long rowBase, int k0, int kh, int K) {
    float tmp[16];
    if (k0 + 32 <= K) {
        const float* wp = W + rowBase + k0 + kh;
        if ((K & 3) == 0) {                   // 16B-aligned fast path (all dense GEMMs)
            const float4* wp4 = (const float4*)wp;
#pragma unroll
            for (int q = 0; q < 4; ++q) {
                float4 f = wp4[q];
                tmp[4 * q + 0] = f.x; tmp[4 * q + 1] = f.y;
                tmp[4 * q + 2] = f.z; tmp[4 * q + 3] = f.w;
            }
        } else {                              // ragged-K (layer 1), unguarded scalar
#pragma unroll
            for (int t = 0; t < 16; ++t) tmp[t] = wp[t];
        }
    } else {                                  // K tail: guarded
#pragma unroll
        for (int t = 0; t < 16; ++t) {
            int gk = k0 + kh + t;
            tmp[t] = (gk < K) ? W[rowBase + gk] : 0.0f;
        }
    }
    v16bf b;
#pragma unroll
    for (int j = 0; j < 8; ++j) {
        b[2 * j]     = (__bf16)tmp[2 * j];
        b[2 * j + 1] = (__bf16)tmp[2 * j + 1];
    }
    return b;
}

// ---------------------------------------------------------------------------
// Generic batched GEMM: C[b][m][n] = act(A[b][m][:] . W[b][n][:] + bias[b][n])
// M multiple of 64, N multiple of 128 (grid.x = N/128), K arbitrary (zero pad).
// 256 threads = 8 waves; wave -> 16-col strip; 4 WMMA accs cover 64 rows.
// ---------------------------------------------------------------------------
__global__ __launch_bounds__(256)
void gemm_bf16_wmma(const float* __restrict__ Aall, long long aStride,
                    const float* __restrict__ Wall, long long wStride,
                    const float* __restrict__ biasAll, long long bStride,
                    float* __restrict__ Call, long long cStride,
                    int M, int N, int K, int relu)
{
    const float* A    = Aall    + (long long)blockIdx.z * aStride;
    const float* W    = Wall    + (long long)blockIdx.z * wStride;
    const float* bias = biasAll + (long long)blockIdx.z * bStride;
    float*       C    = Call    + (long long)blockIdx.z * cStride;

    __shared__ unsigned int AswU[A_LDS_DWORDS];
    const int wave    = threadIdx.x >> 5;
    const int lane    = threadIdx.x & 31;
    const int mbase   = blockIdx.y * 64;
    const int colbase = blockIdx.x * 128 + wave * 16;
    const int n       = colbase + (lane & 15);
    const int kh      = (lane & 16) ? 16 : 0;
    const long long rowBase = (long long)n * K;

    v8f acc[4] = {};

    for (int k0 = 0; k0 < K; k0 += 32) {
        if (k0 + 32 <= K && (K & 3) == 0) {
            // Fast path: straight-line, vectorized, no guards. 512 quads.
#pragma unroll
            for (int it = 0; it < 2; ++it) {
                int p = (int)threadIdx.x + it * 256;
                int r = p >> 3;
                int c = (p & 7) << 2;
                const float4 f = *(const float4*)(A + (long long)(mbase + r) * K + (k0 + c));
                stage_quad(AswU, r, c, f.x, f.y, f.z, f.w);
            }
        } else {
            for (int p = threadIdx.x; p < 1024; p += 256) {
                int r  = p >> 4;
                int c  = (p & 15) << 1;
                int gk = k0 + c;
                const float* arow = A + (long long)(mbase + r) * K;
                float f0 = (gk     < K) ? arow[gk]     : 0.0f;
                float f1 = (gk + 1 < K) ? arow[gk + 1] : 0.0f;
                stage_pair(AswU, r, c, f0, f1);
            }
        }
        __syncthreads();

        if (k0 + 32 < K) __builtin_prefetch(&W[rowBase + k0 + 32], 0, 1);

        v16bf bfrag = make_b_frag(W, rowBase, k0, kh, K);
#pragma unroll
        for (int mb = 0; mb < 4; ++mb) {
            v16bf afrag = read_a_frag(AswU, lane, mb);
            acc[mb] = __builtin_amdgcn_wmma_f32_16x16x32_bf16(
                false, afrag, false, bfrag, (short)0, acc[mb], false, false);
        }
        __syncthreads();
    }

    const float bv = bias[n];
#pragma unroll
    for (int mb = 0; mb < 4; ++mb) {
#pragma unroll
        for (int r = 0; r < 8; ++r) {
            int m = mbase + mb * 16 + ((lane & 16) ? 8 : 0) + r;
            float v = acc[mb][r] + bv;
            if (relu) v = fmaxf(v, 0.0f);
            C[(long long)m * N + n] = v;
        }
    }
}

// ---------------------------------------------------------------------------
// Layer 1: per-label gather-GEMM. h1[l][b][g] = relu(image[b, idx[l]] . W1[l][g] + b1)
// grid = (4 colgroups of 128, 33 labels); streams W1 (~440 MB) exactly once.
// ---------------------------------------------------------------------------
__global__ __launch_bounds__(256)
void layer1_wmma(const float* __restrict__ image, L1Args args, float* __restrict__ H1)
{
    const int l       = blockIdx.y;
    const int* idx    = args.idx[l];
    const float* W    = args.w1[l];
    const float* bias = args.b1[l];
    const int K       = args.n[l];

    __shared__ unsigned int AswU[A_LDS_DWORDS];
    const int wave    = threadIdx.x >> 5;
    const int lane    = threadIdx.x & 31;
    const int colbase = blockIdx.x * 128 + wave * 16;
    const int n       = colbase + (lane & 15);
    const int kh      = (lane & 16) ? 16 : 0;
    const long long rowBase = (long long)n * K;

    v8f acc[4] = {};

    for (int k0 = 0; k0 < K; k0 += 32) {
        if (k0 + 32 <= K) {
            // Fast path: unguarded gather, fully unrolled.
#pragma unroll
            for (int it = 0; it < 2; ++it) {
                int p = (int)threadIdx.x + it * 256;
                int r = p >> 3;
                int c = (p & 7) << 2;
                const int4 ci = *(const int4*)(idx + k0 + c);
                const float* irow = image + (long long)r * VWIDTH;
                stage_quad(AswU, r, c, irow[ci.x], irow[ci.y], irow[ci.z], irow[ci.w]);
            }
        } else {
            for (int p = threadIdx.x; p < 1024; p += 256) {
                int r  = p >> 4;
                int c  = (p & 15) << 1;
                int gk = k0 + c;
                int col0 = (gk     < K) ? idx[gk]     : -1;
                int col1 = (gk + 1 < K) ? idx[gk + 1] : -1;
                const float* irow = image + (long long)r * VWIDTH;
                float f0 = (col0 >= 0) ? irow[col0] : 0.0f;
                float f1 = (col1 >= 0) ? irow[col1] : 0.0f;
                stage_pair(AswU, r, c, f0, f1);
            }
        }
        __syncthreads();

        if (k0 + 32 < K) __builtin_prefetch(&W[rowBase + k0 + 32], 0, 1);

        v16bf bfrag = make_b_frag(W, rowBase, k0, kh, K);
#pragma unroll
        for (int mb = 0; mb < 4; ++mb) {
            v16bf afrag = read_a_frag(AswU, lane, mb);
            acc[mb] = __builtin_amdgcn_wmma_f32_16x16x32_bf16(
                false, afrag, false, bfrag, (short)0, acc[mb], false, false);
        }
        __syncthreads();
    }

    float* C = H1 + (long long)l * 64 * 512;
    const float bv = bias[n];
#pragma unroll
    for (int mb = 0; mb < 4; ++mb) {
#pragma unroll
        for (int r = 0; r < 8; ++r) {
            int m = mb * 16 + ((lane & 16) ? 8 : 0) + r;
            C[(long long)m * 512 + n] = fmaxf(acc[mb][r] + bv, 0.0f);
        }
    }
}

// ---------------------------------------------------------------------------
// Attention: one block per (b,h); S=33, HD=32. Tiny -> VALU + LDS.
// QKV: [2112][384] (q|k|v each 128). CTX: [2112][128].
// ---------------------------------------------------------------------------
__global__ __launch_bounds__(256)
void attn_kernel(const float* __restrict__ QKV, float* __restrict__ CTX)
{
    const int b = blockIdx.x >> 2;
    const int h = blockIdx.x & 3;
    __shared__ float q[33 * 32], k[33 * 32], v[33 * 32], s[33 * 34];
    const int tid = threadIdx.x;

    for (int p = tid; p < 33 * 32; p += 256) {
        int i = p >> 5, d = p & 31;
        long long base = (long long)(i * 64 + b) * 384 + h * 32 + d;
        q[p] = QKV[base];
        k[p] = QKV[base + 128];
        v[p] = QKV[base + 256];
    }
    __syncthreads();

    const float scale = 0.17677669529663687f;  // 1/sqrt(32)
    for (int p = tid; p < 33 * 33; p += 256) {
        int i = p / 33, j = p % 33;
        float acc = 0.0f;
        for (int d = 0; d < 32; ++d) acc += q[i * 32 + d] * k[j * 32 + d];
        s[i * 34 + j] = acc * scale;
    }
    __syncthreads();

    if (tid < 33) {
        float m = -1e30f;
        for (int j = 0; j < 33; ++j) m = fmaxf(m, s[tid * 34 + j]);
        float sum = 0.0f;
        for (int j = 0; j < 33; ++j) {
            float e = __expf(s[tid * 34 + j] - m);
            s[tid * 34 + j] = e;
            sum += e;
        }
        float inv = 1.0f / sum;
        for (int j = 0; j < 33; ++j) s[tid * 34 + j] *= inv;
    }
    __syncthreads();

    for (int p = tid; p < 33 * 32; p += 256) {
        int i = p >> 5, d = p & 31;
        float acc = 0.0f;
        for (int j = 0; j < 33; ++j) acc += s[i * 34 + j] * v[j * 32 + d];
        CTX[(long long)(i * 64 + b) * 128 + h * 32 + d] = acc;
    }
}

// ---------------------------------------------------------------------------
// x = LayerNorm(x + res) over D=128; one wave per token, 4 elems per lane.
// ---------------------------------------------------------------------------
__global__ __launch_bounds__(256)
void add_ln_kernel(float* __restrict__ x, const float* __restrict__ res,
                   const float* __restrict__ w, const float* __restrict__ b)
{
    const int wave = threadIdx.x >> 5;
    const int lane = threadIdx.x & 31;
    const int t    = blockIdx.x * 8 + wave;   // 264 * 8 = 2112 tokens
    const long long base = (long long)t * 128 + lane * 4;

    float vv[4];
    float sum = 0.0f;
#pragma unroll
    for (int j = 0; j < 4; ++j) { vv[j] = x[base + j] + res[base + j]; sum += vv[j]; }
#pragma unroll
    for (int off = 16; off > 0; off >>= 1) sum += __shfl_xor(sum, off, 32);
    const float mean = sum * (1.0f / 128.0f);

    float var = 0.0f;
#pragma unroll
    for (int j = 0; j < 4; ++j) { float d = vv[j] - mean; var += d * d; }
#pragma unroll
    for (int off = 16; off > 0; off >>= 1) var += __shfl_xor(var, off, 32);
    var *= (1.0f / 128.0f);
    const float r = rsqrtf(var + 1e-5f);

#pragma unroll
    for (int j = 0; j < 4; ++j)
        x[base + j] = (vv[j] - mean) * r * w[lane * 4 + j] + b[lane * 4 + j];
}

// ---------------------------------------------------------------------------
// pooled = mean over S of x; out = pooled @ fc_w.T + fc_b.  192 outputs.
// ---------------------------------------------------------------------------
__global__ __launch_bounds__(192)
void pool_fc_kernel(const float* __restrict__ x, const float* __restrict__ fcw,
                    const float* __restrict__ fcb, float* __restrict__ out)
{
    const int tid = threadIdx.x;
    if (tid >= 192) return;
    const int b = tid / 3, c = tid % 3;
    float acc = fcb[c];
    for (int d = 0; d < 128; ++d) {
        float p = 0.0f;
        for (int s = 0; s < 33; ++s) p += x[(long long)(s * 64 + b) * 128 + d];
        acc += (p * (1.0f / 33.0f)) * fcw[c * 128 + d];
    }
    out[b * 3 + c] = acc;
}

// ---------------------------------------------------------------------------
extern "C" void kernel_launch(void* const* d_in, const int* in_sizes, int n_in,
                              void* d_out, int out_size, void* d_ws, size_t ws_size,
                              hipStream_t stream)
{
    (void)n_in; (void)out_size; (void)ws_size;

    const float* image = (const float*)d_in[0];
    L1Args ra;
    for (int l = 0; l < NLAB; ++l) {
        ra.idx[l] = (const int*)  d_in[1 + l];
        ra.w1[l]  = (const float*)d_in[34 + l];
        ra.b1[l]  = (const float*)d_in[67 + l];
        ra.n[l]   = in_sizes[1 + l];
    }
    const float* W2    = (const float*)d_in[100];
    const float* b2    = (const float*)d_in[101];
    const float* W3    = (const float*)d_in[102];
    const float* b3    = (const float*)d_in[103];
    const float* qkv_w = (const float*)d_in[104];
    const float* qkv_b = (const float*)d_in[105];
    const float* out_w = (const float*)d_in[106];
    const float* out_b = (const float*)d_in[107];
    const float* ln1_w = (const float*)d_in[108];
    const float* ln1_b = (const float*)d_in[109];
    const float* ff1_w = (const float*)d_in[110];
    const float* ff1_b = (const float*)d_in[111];
    const float* ff2_w = (const float*)d_in[112];
    const float* ff2_b = (const float*)d_in[113];
    const float* ln2_w = (const float*)d_in[114];
    const float* ln2_b = (const float*)d_in[115];
    const float* fc_w  = (const float*)d_in[116];
    const float* fc_b  = (const float*)d_in[117];

    float* ws   = (float*)d_ws;
    float* H1   = ws;                    // 33*64*512 = 1081344 f
    float* H2   = ws + 1081344;          // 33*64*256 =  540672 f
    float* X    = ws + 1622016;          // 33*64*128 =  270336 f  (token-major [s*64+b][128])
    float* FF1  = ws + 1892352;          // 2112*512  = 1081344 f
    float* QKV  = H1;                    // 2112*384 (H1 dead after layer 2)
    float* CTX  = H2;                    // 2112*128 (H2 dead after layer 3)
    float* PROJ = H2 + 270336;           // 2112*128
    float* FF2  = PROJ;                  // PROJ dead after first add_ln

    // Stage 1: ragged gather + expert MLP layer 1 -> H1 [33][64][512]
    layer1_wmma<<<dim3(4, 33, 1), 256, 0, stream>>>(image, ra, H1);

    // Stage 2/3: batched expert layers -> H2 [33][64][256], X [33][64][128]
    gemm_bf16_wmma<<<dim3(2, 1, 33), 256, 0, stream>>>(
        H1, 64LL * 512, W2, 256LL * 512, b2, 256LL, H2, 64LL * 256, 64, 256, 512, 1);
    gemm_bf16_wmma<<<dim3(1, 1, 33), 256, 0, stream>>>(
        H2, 64LL * 256, W3, 128LL * 256, b3, 128LL, X, 64LL * 128, 64, 128, 256, 0);

    // Stage 4: 4 post-norm transformer layers over 2112 tokens (M = 33*64)
    for (int i = 0; i < 4; ++i) {
        gemm_bf16_wmma<<<dim3(3, 33, 1), 256, 0, stream>>>(
            X, 0, qkv_w + (long long)i * 384 * 128, 0, qkv_b + i * 384, 0,
            QKV, 0, 2112, 384, 128, 0);
        attn_kernel<<<dim3(256), 256, 0, stream>>>(QKV, CTX);
        gemm_bf16_wmma<<<dim3(1, 33, 1), 256, 0, stream>>>(
            CTX, 0, out_w + (long long)i * 128 * 128, 0, out_b + i * 128, 0,
            PROJ, 0, 2112, 128, 128, 0);
        add_ln_kernel<<<dim3(264), 256, 0, stream>>>(X, PROJ, ln1_w + i * 128, ln1_b + i * 128);
        gemm_bf16_wmma<<<dim3(4, 33, 1), 256, 0, stream>>>(
            X, 0, ff1_w + (long long)i * 512 * 128, 0, ff1_b + i * 512, 0,
            FF1, 0, 2112, 512, 128, 1);
        gemm_bf16_wmma<<<dim3(1, 33, 1), 256, 0, stream>>>(
            FF1, 0, ff2_w + (long long)i * 128 * 512, 0, ff2_b + i * 128, 0,
            FF2, 0, 2112, 128, 512, 0);
        add_ln_kernel<<<dim3(264), 256, 0, stream>>>(X, FF2, ln2_w + i * 128, ln2_b + i * 128);
    }

    // Stage 5: mean-pool over S and final FC -> [64][3]
    pool_fc_kernel<<<dim3(1), 192, 0, stream>>>(X, fc_w, fc_b, (float*)d_out);
}
